// InterGate_52286931862183
// MI455X (gfx1250) — matrix operand order repaired and checked by reading, not compiled, not run on hardware
//
#include <hip/hip_runtime.h>
#include <hip/hip_bf16.h>

// Problem constants (reference: B=1, C=192, H=W=96)
#define C_DIM 192
#define HW    9216      // 96*96
#define QTILE 16        // queries per wave
#define KBLK  32        // keys per iteration
#define WAVES 8         // waves per block (share K/V tiles via LDS)
#define NIT   (HW / KBLK)

// LDS buffer geometry (TDM pads inserted for bank-conflict avoidance)
//  K tile: 32 rows x 192ch bf16 = 12288B data, +8B pad per 128B -> 13056B, row stride 408B
//  V tile: 192 rows x 32key bf16 = 12288B data, +16B pad per 64B -> 15360B, row stride 80B
#define KBUF_SZ 13056
#define VBUF_SZ 15360
#define K_OFF(buf) ((buf) * KBUF_SZ)
#define V_OFF(buf) (2 * KBUF_SZ + (buf) * VBUF_SZ)
#define P_OFF      (2 * KBUF_SZ + 2 * VBUF_SZ)
#define LDS_TOTAL  (P_OFF + WAVES * QTILE * KBLK * 2)   // 65,024 B

typedef __attribute__((ext_vector_type(16))) __bf16 v16bf;
typedef __attribute__((ext_vector_type(8)))  float  v8f;
typedef __attribute__((ext_vector_type(4)))  unsigned int v4u;
typedef __attribute__((ext_vector_type(8)))  int v8i;
typedef __attribute__((ext_vector_type(4)))  int v4i;

union FragBF { v16bf v; uint4 q[2]; };

__device__ __forceinline__ unsigned short f2bf(float f) {
  union { float f; unsigned u; } x; x.f = f;
  unsigned u = x.u;
  unsigned r = (u + 0x7FFFu + ((u >> 16) & 1u)) >> 16;   // RNE fp32->bf16
  return (unsigned short)r;
}

// ---------------------------------------------------------------------------
// TDM: issue a 2D bf16 tensor_load_to_lds per ISA 08_async_tensor D# layout.
//   group0: [1:0]=count=1, [63:32]=lds_addr, [120:64]=global_addr, [127:126]=type2
//   group1: data_size=1(2B), pad ctl, tensor_dim0/1, tile_dim0/1, dim0 stride
// This toolchain exposes the 6-arg builtin: (v4u, v8i, v4i, v4i, v8i, i32 cpol);
// the 2D case needs only the first two groups, rest zero-filled.
// ---------------------------------------------------------------------------
__device__ __forceinline__ void tdm_load_2d(unsigned lds_addr, unsigned long long gaddr,
                                            unsigned tile_d0, unsigned tile_d1,
                                            unsigned tensor_d0, unsigned tensor_d1,
                                            unsigned long long stride0,
                                            unsigned pad_int_code, unsigned pad_amt_code,
                                            bool pad_en) {
  v4u g0;
  g0[0] = 1u;                                             // count=1, no gather
  g0[1] = lds_addr;
  g0[2] = (unsigned)(gaddr & 0xffffffffull);
  g0[3] = (unsigned)((gaddr >> 32) & 0x01ffffffull) | (2u << 30);
  v8i g1;
  g1[0] = (int)((1u << 16)                                // data_size = 2 bytes
              | (pad_en ? (1u << 20) : 0u)
              | (pad_int_code << 22)
              | (pad_amt_code << 25));
  g1[1] = (int)((tensor_d0 & 0xffffu) << 16);             // atomic_barrier_addr=0
  g1[2] = (int)(((tensor_d0 >> 16) & 0xffffu) | ((tensor_d1 & 0xffffu) << 16));
  g1[3] = (int)(((tensor_d1 >> 16) & 0xffffu) | ((tile_d0 & 0xffffu) << 16));
  g1[4] = (int)(tile_d1 & 0xffffu);                       // tile_dim2 = 0
  g1[5] = (int)(unsigned)(stride0 & 0xffffffffull);
  g1[6] = (int)(unsigned)((stride0 >> 32) & 0xffffull);   // tensor_dim1_stride = 0
  g1[7] = 0;
  v4i g2 = (v4i){0, 0, 0, 0};
  v4i g3 = (v4i){0, 0, 0, 0};
  v8i g4 = (v8i){0, 0, 0, 0, 0, 0, 0, 0};
  __builtin_amdgcn_tensor_load_to_lds(g0, g1, g2, g3, g4, 0);
}

// ---------------------------------------------------------------------------
// Prep: bf16 operands.  Qb/Kb row-major [N,C]; Vb = lat in native [C,N] layout.
// ---------------------------------------------------------------------------
__global__ void prep_kernel(const float* __restrict__ latent,
                            const float* __restrict__ refined,
                            const float* __restrict__ w4,
                            unsigned short* __restrict__ Qb,
                            unsigned short* __restrict__ Kb,
                            unsigned short* __restrict__ Vb) {
  int idx = blockIdx.x * blockDim.x + threadIdx.x;
  if (idx >= C_DIM * HW) return;
  int c = idx / HW;
  int n = idx - c * HW;
  float la = latent[idx] * w4[idx];
  unsigned short lb = f2bf(la);
  Vb[idx]           = lb;
  Qb[n * C_DIM + c] = lb;
  Kb[n * C_DIM + c] = f2bf(refined[idx]);
}

// ---------------------------------------------------------------------------
// Flash attention: 8 waves/block, 16 queries/wave; K/V tiles double-buffered
// in LDS via TDM; bf16 WMMA with fp32 accumulation and online softmax.
// ---------------------------------------------------------------------------
__launch_bounds__(WAVES * 32, 1)
__global__ void flash_kernel(const unsigned short* __restrict__ Qb,
                             const unsigned short* __restrict__ Kb,
                             const unsigned short* __restrict__ Vb,
                             const float* __restrict__ refined,
                             const float* __restrict__ w5,
                             float* __restrict__ out) {
  __shared__ __align__(16) unsigned char lds[LDS_TOTAL];

  const int lane   = threadIdx.x & 31;
  const int wave   = threadIdx.x >> 5;
  const int qbase  = (blockIdx.x * WAVES + wave) * QTILE;
  const int lane16 = lane & 15;
  const int half   = lane >> 4;
  const int selA   = half * 16;    // A-layout: elem i -> K = selA + i
  const unsigned ldsAbs = (unsigned)(size_t)(&lds[0]);
  unsigned short* myP = (unsigned short*)(lds + P_OFF + wave * (QTILE * KBLK * 2));

  // ---- Q A-fragments (6 chunks of 32 channels), straight from global
  FragBF qf[6];
  {
    const unsigned short* qrow = Qb + (size_t)(qbase + lane16) * C_DIM;
#pragma unroll
    for (int ch = 0; ch < 6; ++ch) {
      const unsigned short* p = qrow + ch * 32 + selA;
      qf[ch].q[0] = *(const uint4*)(p);
      qf[ch].q[1] = *(const uint4*)(p + 8);
    }
  }

  v8f O[12];
#pragma unroll
  for (int t = 0; t < 12; ++t) O[t] = (v8f){0.f,0.f,0.f,0.f,0.f,0.f,0.f,0.f};
  float m_run[8], l_run[8];
#pragma unroll
  for (int r = 0; r < 8; ++r) { m_run[r] = -1e30f; l_run[r] = 0.f; }

  // ---- prologue: TDM-load buffer 0 (K tile contiguous, V tile strided gather)
  if (wave == 0) {
    tdm_load_2d(ldsAbs + K_OFF(0), (unsigned long long)(size_t)(Kb),
                /*tile*/ 6144, 1, /*tensor*/ 6144, 1, 6144,
                /*pad: every 128B add 8B*/ 4, 1, true);
    tdm_load_2d(ldsAbs + V_OFF(0), (unsigned long long)(size_t)(Vb),
                /*tile*/ KBLK, C_DIM, /*tensor*/ HW, C_DIM, HW,
                /*pad: every 64B add 16B*/ 3, 3, true);
  }

  int buf = 0;
  for (int it = 0; it < NIT; ++it) {
    const int nb = it * KBLK;

    if (wave == 0) {
      if (it + 1 < NIT) {   // kick off next block into the other buffer
        const int nn = nb + KBLK;
        tdm_load_2d(ldsAbs + K_OFF(buf ^ 1),
                    (unsigned long long)(size_t)(Kb + (size_t)nn * C_DIM),
                    6144, 1, 6144, 1, 6144, 4, 1, true);
        tdm_load_2d(ldsAbs + V_OFF(buf ^ 1),
                    (unsigned long long)(size_t)(Vb + nn),
                    KBLK, C_DIM, HW, C_DIM, HW, 3, 3, true);
        __builtin_amdgcn_s_wait_tensorcnt(2);   // current buffer's 2 ops done
      } else {
        __builtin_amdgcn_s_wait_tensorcnt(0);
      }
    }
    __syncthreads();                            // publish LDS tiles to all waves

    const unsigned kOff = K_OFF(buf);
    const unsigned vOff = V_OFF(buf);

    // ---- S = Q * K^T (two 16x16 tiles) from LDS K fragments
    v8f S0 = (v8f){0.f,0.f,0.f,0.f,0.f,0.f,0.f,0.f};
    v8f S1 = (v8f){0.f,0.f,0.f,0.f,0.f,0.f,0.f,0.f};
#pragma unroll
    for (int ch = 0; ch < 6; ++ch) {
      FragBF kf;
      int b0 = lane16 * 384 + ch * 64 + half * 32;          // data-byte offset
      const uint4* p0 = (const uint4*)(lds + kOff + b0 + ((b0 >> 7) << 3));
      kf.q[0] = p0[0]; kf.q[1] = p0[1];
      S0 = __builtin_amdgcn_wmma_f32_16x16x32_bf16(false, qf[ch].v, false, kf.v,
                                                   (short)0, S0, false, false);
      int b1 = (16 + lane16) * 384 + ch * 64 + half * 32;
      const uint4* p1 = (const uint4*)(lds + kOff + b1 + ((b1 >> 7) << 3));
      kf.q[0] = p1[0]; kf.q[1] = p1[1];
      S1 = __builtin_amdgcn_wmma_f32_16x16x32_bf16(false, qf[ch].v, false, kf.v,
                                                   (short)0, S1, false, false);
    }

    // ---- online softmax (row r + 8*half on lanes 0-15 / 16-31)
    float scl[8];
#pragma unroll
    for (int r = 0; r < 8; ++r) {
      float mx = fmaxf(S0[r], S1[r]);
#pragma unroll
      for (int off = 1; off <= 8; off <<= 1)
        mx = fmaxf(mx, __shfl_xor(mx, off, 32));
      float mn = fmaxf(m_run[r], mx);
      float sc = __expf(m_run[r] - mn);
      float p0 = __expf(S0[r] - mn);
      float p1 = __expf(S1[r] - mn);
      float rs = p0 + p1;
#pragma unroll
      for (int off = 1; off <= 8; off <<= 1)
        rs += __shfl_xor(rs, off, 32);
      l_run[r] = l_run[r] * sc + rs;
      m_run[r] = mn;
      scl[r]   = sc;
      int row = r + half * 8;                 // C-layout -> LDS 16x32 bf16
      myP[row * KBLK + lane16]      = f2bf(p0);
      myP[row * KBLK + 16 + lane16] = f2bf(p1);
    }
#pragma unroll
    for (int t = 0; t < 12; ++t)
#pragma unroll
      for (int r = 0; r < 8; ++r) O[t][r] *= scl[r];

    __builtin_amdgcn_wave_barrier();          // wave-internal LDS store->load order
    FragBF pf;                                // P back as A-fragment (16x32)
    {
      const unsigned short* pp = myP + lane16 * KBLK + selA;
      pf.q[0] = *(const uint4*)(pp);
      pf.q[1] = *(const uint4*)(pp + 8);
    }
    __builtin_amdgcn_wave_barrier();

    // ---- O += P * V from LDS V fragments (row stride 80B, conflict-free)
#pragma unroll
    for (int t = 0; t < 12; ++t) {
      FragBF vf;
      int c = t * 16 + lane16;
      const uint4* p = (const uint4*)(lds + vOff + c * 80 + half * 32);
      vf.q[0] = p[0]; vf.q[1] = p[1];
      O[t] = __builtin_amdgcn_wmma_f32_16x16x32_bf16(false, pf.v, false, vf.v,
                                                     (short)0, O[t], false, false);
    }

    __syncthreads();                          // all readers done before buffer reuse
    buf ^= 1;
  }

  // ---- epilogue: out[c,m] = refined*w5 + O[m,c]/l[m]
  float invl[8];
#pragma unroll
  for (int r = 0; r < 8; ++r) invl[r] = 1.0f / l_run[r];
#pragma unroll
  for (int t = 0; t < 12; ++t) {
    int c = t * 16 + lane16;
#pragma unroll
    for (int r = 0; r < 8; ++r) {
      int m = qbase + r + half * 8;
      size_t idx = (size_t)c * HW + m;
      out[idx] = refined[idx] * w5[idx] + O[t][r] * invl[r];
    }
  }
}

// ---------------------------------------------------------------------------
extern "C" void kernel_launch(void* const* d_in, const int* in_sizes, int n_in,
                              void* d_out, int out_size, void* d_ws, size_t ws_size,
                              hipStream_t stream) {
  const float* latent  = (const float*)d_in[0];
  const float* refined = (const float*)d_in[1];
  const float* w4      = (const float*)d_in[2];
  const float* w5      = (const float*)d_in[3];
  float* out = (float*)d_out;

  const size_t elems = (size_t)C_DIM * HW;          // 1,769,472
  unsigned short* Qb = (unsigned short*)d_ws;       // [HW, C] bf16
  unsigned short* Kb = Qb + elems;                  // [HW, C] bf16
  unsigned short* Vb = Kb + elems;                  // [C, HW] bf16  (~10.6 MB total)

  prep_kernel<<<(unsigned)((elems + 255) / 256), 256, 0, stream>>>(
      latent, refined, w4, Qb, Kb, Vb);

  const int nblocks = HW / (QTILE * WAVES);         // 72
  flash_kernel<<<nblocks, WAVES * 32, 0, stream>>>(Qb, Kb, Vb, refined, w5, out);
}
